// SharedModule_45243185496736
// MI455X (gfx1250) — matrix operand order repaired and checked by reading, not compiled, not run on hardware
//
#include <hip/hip_runtime.h>
#include <hip/hip_bf16.h>
#include <math.h>

typedef __attribute__((ext_vector_type(16))) _Float16 v16h;
typedef __attribute__((ext_vector_type(8)))  _Float16 h8;
typedef __attribute__((ext_vector_type(4)))  _Float16 h4;
typedef __attribute__((ext_vector_type(8)))  float    v8f;
typedef __attribute__((ext_vector_type(4)))  float    v4f;

union V16H { v16h v; h8 h[2]; };

#define WMMA_F16(A, B, C) \
  __builtin_amdgcn_wmma_f32_16x16x32_f16(false, (A), false, (B), (short)0, (C), false, false)

__device__ __forceinline__ float sigmoidf_(float x) { return 1.0f / (1.0f + expf(-x)); }

// ---------------- prep kernels (run inside the graph, every launch) ----------------

// adj (f32 [4096][4096]) -> adj16 = adj * 4096 (f16), avoids f16 subnormals
__global__ void conv_adj_k(const float* __restrict__ adj, _Float16* __restrict__ adj16) {
  const size_t i = ((size_t)blockIdx.x * blockDim.x + threadIdx.x) * 4;
  v4f v = *(const v4f*)(adj + i);
  h4 o;
#pragma unroll
  for (int j = 0; j < 4; ++j) o[j] = (_Float16)(v[j] * 4096.0f);
  *(h4*)(adj16 + i) = o;
}

// src f32 [K][Ncol] row-major -> dst f16 [Ncol][K]  (B-operand layout: K contiguous)
__global__ void transpose_w_k(const float* __restrict__ src, _Float16* __restrict__ dst,
                              int K, int Ncol) {
  const int idx = blockIdx.x * blockDim.x + threadIdx.x;
  if (idx >= K * Ncol) return;
  const int nn = idx / K, kk = idx % K;
  dst[idx] = (_Float16)src[kk * Ncol + nn];
}

__global__ void zero_hc_k(_Float16* __restrict__ h16, float* __restrict__ c) {
  const int i = blockIdx.x * blockDim.x + threadIdx.x;
  h16[i] = (_Float16)0.0f;
  c[i] = 0.0f;
}

// ---------------- per-step kernel 1: gates + q (all WMMA) ----------------
// gates[n][0:256] = x_t[n] @ W_ih + h[n] @ W_hh + bias   (f32 to ws)
// qT[h][n]        = tanh(h[n] @ W_q + b_q)               (f16, transposed, to ws)
// grid: 64 blocks x 128 threads; each wave owns a 16-row tile (64 rows per WG)
__global__ void __launch_bounds__(128)
k1_gates_q(const float* __restrict__ x,       // [4096][128][32] f32
           const float* __restrict__ bias,    // [256]
           const float* __restrict__ bias_q,  // [64]
           const _Float16* __restrict__ h16,  // [4096][64]
           const _Float16* __restrict__ WihT, // [256][32]
           const _Float16* __restrict__ WhhT, // [256][64]
           const _Float16* __restrict__ WqT,  // [64][64]
           float* __restrict__ gates,         // [4096][256]
           _Float16* __restrict__ qT,         // [64][4096]
           int t) {
  __shared__ _Float16 qlds[64][66];  // pad 66 to break bank conflicts on column reads
  const int lane = threadIdx.x & 31;
  const int wv   = threadIdx.x >> 5;
  const int Rblk = blockIdx.x * 64;
  const int R0   = Rblk + wv * 16;
  const int arow = R0 + (lane & 15);
  const int ka   = (lane >> 4) * 8;   // A-operand K sub-offset
  const int bn   = lane & 15;         // B-operand column within tile
  const int kb   = (lane >> 4) * 8;   // B-operand K base
  const int rb   = (lane >> 4) * 8;   // C/D row sub-offset

  // A_x: x[arow][t][0:32] f32 -> f16, WMMA A layout (K=32)
  const float* xp = x + ((size_t)arow * 128 + t) * 32;
  v4f x0 = *(const v4f*)(xp + ka);
  v4f x1 = *(const v4f*)(xp + ka + 4);
  v4f x2 = *(const v4f*)(xp + ka + 16);
  v4f x3 = *(const v4f*)(xp + ka + 20);
  v16h Ax;
#pragma unroll
  for (int i = 0; i < 4; ++i) {
    Ax[i]      = (_Float16)x0[i];
    Ax[4 + i]  = (_Float16)x1[i];
    Ax[8 + i]  = (_Float16)x2[i];
    Ax[12 + i] = (_Float16)x3[i];
  }
  // A_h: h16[arow][0:64] -> two K=32 A tiles
  const _Float16* hp = h16 + (size_t)arow * 64;
  V16H Ah0, Ah1;
  Ah0.h[0] = *(const h8*)(hp + ka);      Ah0.h[1] = *(const h8*)(hp + 16 + ka);
  Ah1.h[0] = *(const h8*)(hp + 32 + ka); Ah1.h[1] = *(const h8*)(hp + 48 + ka);

  // gates: 16 column tiles of 16
  for (int ct = 0; ct < 16; ++ct) {
    const int col = ct * 16 + bn;
    V16H B1, B2, B3;
    const _Float16* w1 = WihT + (size_t)col * 32 + kb;
    B1.h[0] = *(const h8*)(w1);      B1.h[1] = *(const h8*)(w1 + 16);
    const _Float16* w2 = WhhT + (size_t)col * 64 + kb;
    B2.h[0] = *(const h8*)(w2);      B2.h[1] = *(const h8*)(w2 + 16);
    const _Float16* w3 = w2 + 32;
    B3.h[0] = *(const h8*)(w3);      B3.h[1] = *(const h8*)(w3 + 16);
    v8f acc = {};
    acc = WMMA_F16(Ax,    B1.v, acc);
    acc = WMMA_F16(Ah0.v, B2.v, acc);
    acc = WMMA_F16(Ah1.v, B3.v, acc);
    const float bc = bias[col];
#pragma unroll
    for (int j = 0; j < 8; ++j)
      gates[(size_t)(R0 + rb + j) * 256 + col] = acc[j] + bc;
  }

  // q = tanh(h @ W_q + b_q): 4 column tiles
#pragma unroll
  for (int ct = 0; ct < 4; ++ct) {
    const int col = ct * 16 + bn;
    V16H B1, B2;
    const _Float16* w1 = WqT + (size_t)col * 64 + kb;
    B1.h[0] = *(const h8*)(w1);      B1.h[1] = *(const h8*)(w1 + 16);
    const _Float16* w2 = w1 + 32;
    B2.h[0] = *(const h8*)(w2);      B2.h[1] = *(const h8*)(w2 + 16);
    v8f acc = {};
    acc = WMMA_F16(Ah0.v, B1.v, acc);
    acc = WMMA_F16(Ah1.v, B2.v, acc);
    const float bq = bias_q[col];
#pragma unroll
    for (int j = 0; j < 8; ++j)
      qlds[wv * 16 + rb + j][col] = (_Float16)tanhf(acc[j] + bq);
  }
  __syncthreads();

  // transposed, coalesced-ish f16 writeout: qT[col][Rblk + r]
  const int cc = threadIdx.x & 63;
  const int rh = threadIdx.x >> 6;
  _Float16* qp = qT + (size_t)cc * 4096 + Rblk + rh * 32;
#pragma unroll
  for (int b = 0; b < 4; ++b) {
    h8 tmp;
#pragma unroll
    for (int i = 0; i < 8; ++i) tmp[i] = qlds[rh * 32 + b * 8 + i][cc];
    *(h8*)(qp + b * 8) = tmp;
  }
}

// ---------------- per-step kernel 2: P = adj @ q (WMMA) + fused LSTM update ----------------
// grid: 128 blocks x 128 threads; block owns 32 rows (2 row tiles per wave, so each
// B tile feeds 2 WMMAs), 4 waves split K=4096, partials reduced in LDS, then fused
// elementwise c/h update + output store.
__global__ void __launch_bounds__(128)
k2_adj_update(const _Float16* __restrict__ adj16,  // [4096][4096], pre-scaled x4096
              const _Float16* __restrict__ qT,     // [64][4096]
              const float* __restrict__ gates,     // [4096][256]
              float* __restrict__ cbuf,            // [4096][64]
              _Float16* __restrict__ h16,          // [4096][64]
              float* __restrict__ out,             // [4096][128][64]
              int t) {
  __shared__ float plds[4][32][64];  // 32 KB
  const int lane = threadIdx.x & 31;
  const int wv   = threadIdx.x >> 5;
  const int R0   = blockIdx.x * 32;
  const int ka   = (lane >> 4) * 8;
  const int bn   = lane & 15;
  const int kb   = (lane >> 4) * 8;

  const _Float16* ap0 = adj16 + (size_t)(R0 + (lane & 15)) * 4096;       // row tile 0
  const _Float16* ap1 = adj16 + (size_t)(R0 + 16 + (lane & 15)) * 4096;  // row tile 1
  const _Float16* qb  = qT + (size_t)bn * 4096 + kb;

  v8f acc[2][4] = {};
  const int kstart = wv * 1024;  // K split across 4 waves
  for (int ks = 0; ks < 32; ++ks) {
    const int k = kstart + ks * 32;
    V16H A0, A1;
    A0.h[0] = *(const h8*)(ap0 + k + ka);
    A0.h[1] = *(const h8*)(ap0 + k + 16 + ka);
    A1.h[0] = *(const h8*)(ap1 + k + ka);
    A1.h[1] = *(const h8*)(ap1 + k + 16 + ka);
#pragma unroll
    for (int ct = 0; ct < 4; ++ct) {
      V16H B;
      const _Float16* q = qb + (size_t)ct * 16 * 4096 + k;
      B.h[0] = *(const h8*)(q);
      B.h[1] = *(const h8*)(q + 16);
      acc[0][ct] = WMMA_F16(A0.v, B.v, acc[0][ct]);  // B reused across both row tiles
      acc[1][ct] = WMMA_F16(A1.v, B.v, acc[1][ct]);
    }
  }

  const int rb = (lane >> 4) * 8;
#pragma unroll
  for (int rt = 0; rt < 2; ++rt) {
#pragma unroll
    for (int ct = 0; ct < 4; ++ct) {
#pragma unroll
      for (int j = 0; j < 8; ++j)
        plds[wv][rt * 16 + rb + j][ct * 16 + bn] = acc[rt][ct][j];
    }
  }
  __syncthreads();

  // fused elementwise: thread -> (row r, 16 columns)
  const int r  = threadIdx.x >> 2;        // 0..31
  const int c0 = (threadIdx.x & 3) * 16;  // 0/16/32/48
  const int n  = R0 + r;
  const float* grow = gates + (size_t)n * 256;
  float* crow    = cbuf + (size_t)n * 64;
  _Float16* hrow = h16 + (size_t)n * 64;
  float* orow    = out + ((size_t)n * 128 + t) * 64;
#pragma unroll
  for (int u = 0; u < 16; ++u) {
    const int cc = c0 + u;
    const float P = (plds[0][r][cc] + plds[1][r][cc] + plds[2][r][cc] + plds[3][r][cc])
                    * (1.0f / 4096.0f);  // undo adj16 pre-scale
    const float it = sigmoidf_(grow[cc]);
    const float ft = sigmoidf_(grow[64 + cc]);
    const float gt = tanhf(grow[128 + cc]);
    const float ot = sigmoidf_(grow[192 + cc]);
    const float cold = crow[cc];
    const float cnew = ft * (cold + P) + it * gt;
    const float hnew = ot * tanhf(cnew);
    crow[cc] = cnew;
    hrow[cc] = (_Float16)hnew;
    orow[cc] = hnew;
  }
}

// ---------------- launch ----------------
extern "C" void kernel_launch(void* const* d_in, const int* in_sizes, int n_in,
                              void* d_out, int out_size, void* d_ws, size_t ws_size,
                              hipStream_t stream) {
  const float* x    = (const float*)d_in[0];  // [4096][128][32]
  const float* adj  = (const float*)d_in[1];  // [4096][4096]
  const float* Wih  = (const float*)d_in[2];  // [32][256]
  const float* Whh  = (const float*)d_in[3];  // [64][256]
  const float* bias = (const float*)d_in[4];  // [256]
  const float* Wq   = (const float*)d_in[5];  // [64][64]
  const float* bq   = (const float*)d_in[6];  // [64]
  float* out = (float*)d_out;                 // [4096][128][64]

  char* w = (char*)d_ws;
  size_t off = 0;
  _Float16* adj16 = (_Float16*)(w + off); off += (size_t)4096 * 4096 * 2;  // 32 MB
  _Float16* h16   = (_Float16*)(w + off); off += (size_t)4096 * 64 * 2;    // 512 KB
  float*    cbuf  = (float*)(w + off);    off += (size_t)4096 * 64 * 4;    // 1 MB
  float*    gates = (float*)(w + off);    off += (size_t)4096 * 256 * 4;   // 4 MB
  _Float16* qT    = (_Float16*)(w + off); off += (size_t)64 * 4096 * 2;    // 512 KB
  _Float16* WihT  = (_Float16*)(w + off); off += (size_t)256 * 32 * 2;
  _Float16* WhhT  = (_Float16*)(w + off); off += (size_t)256 * 64 * 2;
  _Float16* WqT   = (_Float16*)(w + off); off += (size_t)64 * 64 * 2;

  conv_adj_k<<<16384, 256, 0, stream>>>(adj, adj16);
  transpose_w_k<<<(32 * 256 + 255) / 256, 256, 0, stream>>>(Wih, WihT, 32, 256);
  transpose_w_k<<<(64 * 256 + 255) / 256, 256, 0, stream>>>(Whh, WhhT, 64, 256);
  transpose_w_k<<<(64 * 64 + 255) / 256, 256, 0, stream>>>(Wq, WqT, 64, 64);
  zero_hc_k<<<(4096 * 64) / 256, 256, 0, stream>>>(h16, cbuf);

  for (int t = 0; t < 128; ++t) {
    k1_gates_q<<<64, 128, 0, stream>>>(x, bias, bq, h16, WihT, WhhT, WqT, gates, qT, t);
    k2_adj_update<<<128, 128, 0, stream>>>(adj16, qT, gates, cbuf, h16, out, t);
  }
}
